// MultiCNNLSTM_25082609009015
// MI455X (gfx1250) — matrix operand order repaired and checked by reading, not compiled, not run on hardware
//
#include <hip/hip_runtime.h>
#include <math.h>

// ---------------------------------------------------------------------------
// MI455X (gfx1250) fused MultiCNN+LSTM encoder.
// Dominant cost: acoustic conv-as-GEMM (~319 GFLOP) -> v_wmma_f32_16x16x32_bf16.
// Inputs are (BT, W, Cin) row-major with Cin fastest, so each conv output row
// is a CONTIGUOUS window of length K*Cin: im2col is free (pointer offsets).
// v2: whole sample staged to LDS once (fits in 320KB WGP LDS), barrier-free
// WMMA k-loop, B-fragment reuse across row tiles, global_prefetch for B.
// ---------------------------------------------------------------------------

typedef __attribute__((ext_vector_type(16))) __bf16 v16bf;
typedef __attribute__((ext_vector_type(8)))  __bf16 v8bf;
typedef __attribute__((ext_vector_type(4)))  __bf16 v4bf;
typedef __attribute__((ext_vector_type(8)))  float  v8f;

#define BT_TOT 1536   // B*T = 32*48
#define OUT_W  448    // 128 + 256 + 64

__device__ __forceinline__ unsigned short f2bf(float f) {
    union { float fp; unsigned int u; } v; v.fp = f;
    unsigned int u = v.u;
    u += 0x7FFFu + ((u >> 16) & 1u);          // round-to-nearest-even
    return (unsigned short)(u >> 16);
}

__device__ __forceinline__ float sigmoidf_fast(float x) {
    return 1.0f / (1.0f + __expf(-x));
}

// A-fragment for V_WMMA_F32_16X16X32_BF16, contiguous k order (B packed the same):
// lane L: row r=L&15, half h=L>>4; elems 0..15 = K[16h..16h+15]
// 16-byte aligned rowbase variant (LSTM: row stride 160 elems = 320B).
__device__ __forceinline__ v16bf load_afrag16(const unsigned short* rowbase, int h) {
    v8bf lo = *reinterpret_cast<const v8bf*>(rowbase + 16 * h);
    v8bf hi = *reinterpret_cast<const v8bf*>(rowbase + 16 * h + 8);
    return __builtin_shufflevector(lo, hi, 0,1,2,3,4,5,6,7,8,9,10,11,12,13,14,15);
}

// 8-byte aligned variant (conv: row stride = CIN elems, CIN % 4 == 0 -> 8B align).
__device__ __forceinline__ v16bf load_afrag8(const unsigned short* rowbase, int h) {
    v4bf a0 = *reinterpret_cast<const v4bf*>(rowbase + 16 * h);
    v4bf a1 = *reinterpret_cast<const v4bf*>(rowbase + 16 * h + 4);
    v4bf a2 = *reinterpret_cast<const v4bf*>(rowbase + 16 * h + 8);
    v4bf a3 = *reinterpret_cast<const v4bf*>(rowbase + 16 * h + 12);
    v8bf lo = __builtin_shufflevector(a0, a1, 0,1,2,3,4,5,6,7);
    v8bf hi = __builtin_shufflevector(a2, a3, 0,1,2,3,4,5,6,7);
    return __builtin_shufflevector(lo, hi, 0,1,2,3,4,5,6,7,8,9,10,11,12,13,14,15);
}

// ---------------------------------------------------------------------------
// Weight repack: conv w (Cout, Cin, Kw) f32 -> bf16, GEMM-K order k = kw*Cin+ci,
// layout dst[((kb*Cout + n)*32) + kk], zero-padded past KD.
// ---------------------------------------------------------------------------
__global__ void pack_conv_w(const float* __restrict__ src, unsigned short* __restrict__ dst,
                            int Cout, int Cin, int Kw, int KB) {
    int KD = Cin * Kw;
    int total = KB * Cout * 32;
    for (int i = blockIdx.x * blockDim.x + threadIdx.x; i < total;
         i += gridDim.x * blockDim.x) {
        int kk = i & 31;
        int rest = i >> 5;
        int n  = rest % Cout;
        int kb = rest / Cout;
        int k = kb * 32 + kk;
        float v = 0.0f;
        if (k < KD) {
            int kw = k / Cin;
            int ci = k - kw * Cin;
            v = src[((size_t)n * Cin + ci) * Kw + kw];
        }
        dst[i] = f2bf(v);
    }
}

// LSTM: Wcat (148 x 512): rows 0..19 = Wih^T, rows 20..147 = Whh^T, pad to 160.
__global__ void pack_lstm_w(const float* __restrict__ Wih, const float* __restrict__ Whh,
                            unsigned short* __restrict__ dst) {
    const int total = 5 * 512 * 32;
    for (int i = blockIdx.x * blockDim.x + threadIdx.x; i < total;
         i += gridDim.x * blockDim.x) {
        int kk = i & 31;
        int rest = i >> 5;
        int n  = rest % 512;
        int kb = rest / 512;
        int k = kb * 32 + kk;
        float v = 0.0f;
        if (k < 20)        v = Wih[(size_t)n * 20 + k];
        else if (k < 148)  v = Whh[(size_t)n * 128 + (k - 20)];
        dst[i] = f2bf(v);
    }
}

// ---------------------------------------------------------------------------
// Masked LSTM, 16 samples per block, 30 steps. z(16x512) = [x_t|h](16x148) x Wcat.
// Wave w's 4 N-tiles are columns {16w, 128+16w, 256+16w, 384+16w} = i,f,g,o for
// hidden slice [16w, 16w+16) -> gate math is lane-local.
// ---------------------------------------------------------------------------
__global__ __launch_bounds__(256) void lstm_kernel(
    const float* __restrict__ xe,        // (BT, 30, 20)
    const int*   __restrict__ lens,      // (BT,)
    const unsigned short* __restrict__ wpk,  // packed (5,512,32) bf16
    const float* __restrict__ bih, const float* __restrict__ bhh,  // (512,)
    float* __restrict__ hout)            // (BT, 30, 128)
{
    __shared__ __align__(16) unsigned short a_lds[16 * 160];
    __shared__ float h_lds[16 * 128];
    __shared__ float c_lds[16 * 128];
    __shared__ int   len_s[16];

    const int n0   = blockIdx.x * 16;
    const int tid  = threadIdx.x;
    const int w    = tid >> 5;
    const int lane = tid & 31;
    const int r    = lane & 15;
    const int h    = lane >> 4;
    const int hh   = 16 * w + r;         // hidden column this lane owns

    for (int i = tid; i < 16 * 128; i += 256) { h_lds[i] = 0.0f; c_lds[i] = 0.0f; }
    if (tid < 16) len_s[tid] = lens[n0 + tid];

    float bgate[4];
#pragma unroll
    for (int q = 0; q < 4; ++q) bgate[q] = bih[q * 128 + hh] + bhh[q * 128 + hh];

    __syncthreads();

    for (int t = 0; t < 30; ++t) {
        // Build A = [x_t (20) | h_prev (128) | pad (12)] in bf16.
        for (int i = tid; i < 16 * 160; i += 256) {
            int s  = i / 160;
            int kp = i - s * 160;
            float v = 0.0f;
            if (kp < 20)       v = xe[((size_t)(n0 + s) * 30 + t) * 20 + kp];
            else if (kp < 148) v = h_lds[s * 128 + (kp - 20)];
            a_lds[i] = f2bf(v);
        }
        __syncthreads();

        v8f acc[4];
#pragma unroll
        for (int q = 0; q < 4; ++q)
#pragma unroll
            for (int j = 0; j < 8; ++j) acc[q][j] = 0.0f;

#pragma unroll
        for (int kb = 0; kb < 5; ++kb) {
            v16bf af = load_afrag16(a_lds + r * 160 + kb * 32, h);
#pragma unroll
            for (int q = 0; q < 4; ++q) {
                int nt = w + 8 * q;  // N-tile -> gate q, hidden block 16w
                v16bf bf = *reinterpret_cast<const v16bf*>(
                    wpk + ((size_t)(kb * 512 + nt * 16 + r)) * 32 + 16 * h);
                acc[q] = __builtin_amdgcn_wmma_f32_16x16x32_bf16(
                    false, af, false, bf, (short)0, acc[q], false, false);
            }
        }

        // Gate nonlinearity + masked state carry. Lane owns rows {8h+j}, col hh.
#pragma unroll
        for (int j = 0; j < 8; ++j) {
            int s = 8 * h + j;
            float ig = sigmoidf_fast(acc[0][j] + bgate[0]);
            float fg = sigmoidf_fast(acc[1][j] + bgate[1]);
            float gg = tanhf(acc[2][j] + bgate[2]);
            float og = sigmoidf_fast(acc[3][j] + bgate[3]);
            float c_old = c_lds[s * 128 + hh];
            float h_old = h_lds[s * 128 + hh];
            float c2 = fg * c_old + ig * gg;
            float h2 = og * tanhf(c2);
            bool valid = t < len_s[s];
            c_lds[s * 128 + hh] = valid ? c2 : c_old;
            h_lds[s * 128 + hh] = valid ? h2 : h_old;
            hout[((size_t)(n0 + s) * 30 + t) * 128 + hh] = valid ? h2 : 0.0f;
        }
        __syncthreads();
    }
}

// ---------------------------------------------------------------------------
// Fused conv1d -> max over valid width -> +bias -> highway, one block per sample.
// Whole sample staged to LDS as bf16 once; k-loop is barrier-free WMMA.
// A-fragment for output row w at K-chunk kb = LDS window (w*CIN + kb*32).
// ---------------------------------------------------------------------------
template<int CIN, int KW, int W, int WOUT, int COUT, int OUTOFF>
__global__ __launch_bounds__(256) void conv_hw_kernel(
    const float* __restrict__ x,            // (BT, W, CIN)
    const unsigned short* __restrict__ wpk, // packed bf16 (KB, COUT, 32)
    const float* __restrict__ cb,           // (COUT,)
    const float* __restrict__ Wp, const float* __restrict__ bp,
    const float* __restrict__ Wg, const float* __restrict__ bg,
    float* __restrict__ out)                // (BT, 448)
{
    constexpr int KD   = CIN * KW;
    constexpr int KB   = (KD + 31) / 32;
    constexpr int MT   = (WOUT + 15) / 16;
    constexpr int NT   = COUT / 16;
    constexpr int TT   = MT * NT;
    constexpr int TPW  = TT / 8;
    constexpr int NPW  = (NT >= 8) ? (NT / 8) : 1;  // col tiles per wave
    constexpr int MPW  = (NT >= 8) ? MT : 1;        // row tiles per wave
    static_assert(NPW * MPW == TPW, "tile mapping mismatch");
    constexpr int XTOT = W * CIN;                   // sample elements
    constexpr int XV   = XTOT / 4;                  // float4 loads (all divisible)

    __shared__ __align__(16) unsigned short lds_x[XTOT + 32];  // +32 zero tail
    __shared__ float tilemax[TT * 16];
    __shared__ float ylds[COUT];

    const int n    = blockIdx.x;
    const int tid  = threadIdx.x;
    const int w    = tid >> 5;
    const int lane = tid & 31;
    const int r    = lane & 15;
    const int h    = lane >> 4;

    // Stage whole sample (f32 -> bf16) with vectorized loads/stores.
    const float4* xb4 = reinterpret_cast<const float4*>(x + (size_t)n * XTOT);
    for (int i = tid; i < XV; i += 256) {
        float4 v = xb4[i];
        unsigned int lo = (unsigned int)f2bf(v.x) | ((unsigned int)f2bf(v.y) << 16);
        unsigned int hi = (unsigned int)f2bf(v.z) | ((unsigned int)f2bf(v.w) << 16);
        uint2 pk; pk.x = lo; pk.y = hi;
        *reinterpret_cast<uint2*>(&lds_x[i * 4]) = pk;
    }
    if (tid < 32) lds_x[XTOT + tid] = 0;   // zero tail so K in [KD, KB*32) is 0
    __syncthreads();

    v8f acc[TPW];
#pragma unroll
    for (int q = 0; q < TPW; ++q)
#pragma unroll
        for (int j = 0; j < 8; ++j) acc[q][j] = 0.0f;

    // Tile ownership: NT>=8 -> wave w owns cols {w, w+8, ...} x all row tiles;
    // NT<8 (emotient NT=4) -> wave w owns single tile (m = w/NT, nt = w%NT).
    const int mbase  = (NT >= 8) ? 0 : (w / NT);
    const int ntbase = (NT >= 8) ? w : (w % NT);

    for (int kb = 0; kb < KB; ++kb) {
        v16bf afr[MPW];
#pragma unroll
        for (int mm = 0; mm < MPW; ++mm) {
            int row = (mbase + mm) * 16 + r;
            afr[mm] = load_afrag8(lds_x + row * CIN + kb * 32, h);
        }
#pragma unroll
        for (int p = 0; p < NPW; ++p) {
            int nt = ntbase + 8 * p;
            const unsigned short* bptr =
                wpk + ((size_t)(kb * COUT + nt * 16 + r)) * 32 + 16 * h;
            v16bf bf = *reinterpret_cast<const v16bf*>(bptr);
            if (kb + 1 < KB)
                __builtin_prefetch(bptr + (size_t)COUT * 32, 0, 1);
#pragma unroll
            for (int mm = 0; mm < MPW; ++mm)
                acc[p * MPW + mm] = __builtin_amdgcn_wmma_f32_16x16x32_bf16(
                    false, afr[mm], false, bf, (short)0, acc[p * MPW + mm],
                    false, false);
        }
    }

    // Per-lane max over valid rows; rows are split across the lane pair (h=0/1).
#pragma unroll
    for (int q = 0; q < TPW; ++q) {
        int p  = q / MPW;
        int mm = q - p * MPW;
        int m  = mbase + mm;
        int nt = ntbase + 8 * p;
        float mx = -3.4e38f;
#pragma unroll
        for (int j = 0; j < 8; ++j) {
            int row = m * 16 + 8 * h + j;
            if (row < WOUT) mx = fmaxf(mx, acc[q][j]);
        }
        mx = fmaxf(mx, __shfl_xor(mx, 16, 32));
        if (h == 0) tilemax[(m * NT + nt) * 16 + r] = mx;
    }
    __syncthreads();

    if (tid < COUT) {
        int nt  = tid >> 4;
        int col = tid & 15;
        float mx = -3.4e38f;
#pragma unroll
        for (int m = 0; m < MT; ++m)
            mx = fmaxf(mx, tilemax[(m * NT + nt) * 16 + col]);
        ylds[tid] = mx + cb[tid];
    }
    __syncthreads();

    // Highway: p = relu(Wp y + bp), g = sigmoid(Wg y + bg), out = g*p + (1-g)*y.
    if (tid < COUT) {
        float y  = ylds[tid];
        float pa = bp[tid];
        float ga = bg[tid];
        const float* wpr = Wp + (size_t)tid * COUT;
        const float* wgr = Wg + (size_t)tid * COUT;
        for (int j = 0; j < COUT; ++j) {
            float yj = ylds[j];
            pa += wpr[j] * yj;
            ga += wgr[j] * yj;
        }
        float p = fmaxf(pa, 0.0f);
        float g = sigmoidf_fast(ga);
        out[(size_t)n * OUT_W + OUTOFF + tid] = g * p + (1.0f - g) * y;
    }
}

// ---------------------------------------------------------------------------
extern "C" void kernel_launch(void* const* d_in, const int* in_sizes, int n_in,
                              void* d_out, int out_size, void* d_ws, size_t ws_size,
                              hipStream_t stream) {
    (void)in_sizes; (void)n_in; (void)out_size; (void)ws_size;

    const float* x_lin   = (const float*)d_in[0];
    const float* x_aco   = (const float*)d_in[1];
    const float* x_emo   = (const float*)d_in[2];
    const int*   win_len = (const int*)  d_in[3];
    /* d_in[4] = length: unused by the reference body */
    const float* convL_w = (const float*)d_in[5];
    const float* convL_b = (const float*)d_in[6];
    const float* convA_w = (const float*)d_in[7];
    const float* convA_b = (const float*)d_in[8];
    const float* convE_w = (const float*)d_in[9];
    const float* convE_b = (const float*)d_in[10];
    const float* hwL_Wp  = (const float*)d_in[11];
    const float* hwL_bp  = (const float*)d_in[12];
    const float* hwL_Wg  = (const float*)d_in[13];
    const float* hwL_bg  = (const float*)d_in[14];
    const float* hwA_Wp  = (const float*)d_in[15];
    const float* hwA_bp  = (const float*)d_in[16];
    const float* hwA_Wg  = (const float*)d_in[17];
    const float* hwA_bg  = (const float*)d_in[18];
    const float* hwE_Wp  = (const float*)d_in[19];
    const float* hwE_bp  = (const float*)d_in[20];
    const float* hwE_Wg  = (const float*)d_in[21];
    const float* hwE_bg  = (const float*)d_in[22];
    const float* lstm_Wih = (const float*)d_in[23];
    const float* lstm_Whh = (const float*)d_in[24];
    const float* lstm_bih = (const float*)d_in[25];
    const float* lstm_bhh = (const float*)d_in[26];
    float* out = (float*)d_out;

    // Workspace carve (256B aligned): packed bf16 weights + LSTM hidden states.
    unsigned char* base = (unsigned char*)d_ws;
    size_t off = 0;
    auto take = [&](size_t bytes) -> void* {
        void* p = base + off;
        off = (off + bytes + 255) & ~((size_t)255);
        return p;
    };
    const size_t szA = 309ull * 256 * 32 * 2;   // acoustic   KB=ceil(9880/32)=309
    const size_t szL = 47ull  * 128 * 32 * 2;   // linguistic KB=ceil(1500/32)=47
    const size_t szE = 12ull  * 64  * 32 * 2;   // emotient   KB=384/32=12
    const size_t szR = 5ull   * 512 * 32 * 2;   // lstm       KB=ceil(148/32)=5
    unsigned short* pkA = (unsigned short*)take(szA);
    unsigned short* pkL = (unsigned short*)take(szL);
    unsigned short* pkE = (unsigned short*)take(szE);
    unsigned short* pkR = (unsigned short*)take(szR);
    float* hout = (float*)take((size_t)BT_TOT * 30 * 128 * sizeof(float));

    // 1) Weight repacks (bf16, B-fragment layout).
    pack_conv_w<<<(int)((szA / 2 + 255) / 256), 256, 0, stream>>>(convA_w, pkA, 256, 988, 10, 309);
    pack_conv_w<<<(int)((szL / 2 + 255) / 256), 256, 0, stream>>>(convL_w, pkL, 128, 300, 5, 47);
    pack_conv_w<<<(int)((szE / 2 + 255) / 256), 256, 0, stream>>>(convE_w, pkE, 64, 128, 3, 12);
    pack_lstm_w<<<(int)((szR / 2 + 255) / 256), 256, 0, stream>>>(lstm_Wih, lstm_Whh, pkR);

    // 2) Masked LSTM over the emotient windows (16 samples per block).
    lstm_kernel<<<BT_TOT / 16, 256, 0, stream>>>(x_emo, win_len, pkR, lstm_bih, lstm_bhh, hout);

    // 3) Fused conv + maxpool + bias + highway per modality (one block/sample).
    conv_hw_kernel<300, 5, 33, 29, 128, 0><<<BT_TOT, 256, 0, stream>>>(
        x_lin, pkL, convL_b, hwL_Wp, hwL_bp, hwL_Wg, hwL_bg, out);
    conv_hw_kernel<988, 10, 50, 41, 256, 128><<<BT_TOT, 256, 0, stream>>>(
        x_aco, pkA, convA_b, hwA_Wp, hwA_bp, hwA_Wg, hwA_bg, out);
    conv_hw_kernel<128, 3, 30, 28, 64, 384><<<BT_TOT, 256, 0, stream>>>(
        hout, pkE, convE_b, hwE_Wp, hwE_bp, hwE_Wg, hwE_bg, out);
}
